// BCD_GLPLinearA_58342835748977
// MI455X (gfx1250) — matrix operand order repaired and checked by reading, not compiled, not run on hardware
//
#include <hip/hip_runtime.h>

typedef __attribute__((ext_vector_type(16))) _Float16 v16h;
typedef __attribute__((ext_vector_type(8)))  float    v8f;

#define S_LEN 2048
#define D_DIM 64
#define V_SZ  32000

// well tables: MAXN = [2,3,5,8,8,8], cumulative offsets, total K = 34
__device__ __forceinline__ int well_of_k(int k) {
    const int KOFF[7] = {0, 2, 5, 10, 18, 26, 34};
    int i = 0;
    while (i < 5 && k >= KOFF[i + 1]) i++;
    return i;
}

// ---------------- seq_mean: [B,D] mean over S of gathered embedding ----------
__global__ void k_seqmean(const int* __restrict__ ids, const float* __restrict__ emb,
                          float* __restrict__ seq) {
    int b = blockIdx.x, d = threadIdx.x;
    float acc = 0.f;
    for (int s = 0; s < S_LEN; s++)
        acc += emb[(size_t)ids[b * S_LEN + s] * D_DIM + d];
    seq[b * D_DIM + d] = acc / (float)S_LEN;
}

// ---------------- Hermite basis sampled at the S positions -------------------
__global__ void k_basis(float* __restrict__ basis) {
    const int MN[6]   = {2, 3, 5, 8, 8, 8};
    const int KOFF[6] = {0, 2, 5, 10, 18, 26};
    const int SC[6]   = {2, 3, 5, 8, 13, 21};
    int i = blockIdx.x;
    float sigma = (float)SC[i] / 5.0f;
    float inv = 1.0f / (sigma * 1.41421356237309515f);
    int mn = MN[i], ko = KOFF[i];
    const float sqpi = 1.7724538509055159f;
    for (int s = threadIdx.x; s < S_LEN; s += blockDim.x) {
        int idx = (int)((double)s * 255.0 / 2047.0);
        if (idx > 255) idx = 255;
        float x   = -3.0f + 6.0f * (float)idx / 255.0f;
        float xi  = x * inv;
        float env = expf(-0.5f * xi * xi);
        float Hm = 0.0f, H = 1.0f;       // H_{n-1}, H_n
        float pw = 1.0f, fact = 1.0f;    // 2^n, n!
        for (int n = 0; n < mn; n++) {
            float norm = 1.0f / sqrtf(pw * fact * sqpi);
            basis[(ko + n) * S_LEN + s] = norm * H * env;
            float Hn1 = 2.0f * xi * H - 2.0f * (float)n * Hm;
            Hm = H; H = Hn1;
            pw *= 2.0f; fact *= (float)(n + 1);
        }
    }
}

// ---------------- all the small algebra in one block -------------------------
struct SmallParams {
    const float* seq;
    const float* basis;
    float* Cmat;                 // [2][34][64] to ws
    const float* wocc[6];
    const float* bocc[6];
    const float* Smat;           // [6,6,64]
    const float* resE;           // [6,64]
    const float* resW;           // [6,64]
    const float* geow;           // [3,64]
    const float* geob;           // [3]
    float* out_tablet;           // [2,64]
    float* out_geo;              // [2,3]
    float* out_amp;              // [2,6,64]
    float* out_strength;         // [1]
};

__global__ __launch_bounds__(256) void k_small(SmallParams p) {
    __shared__ float s_seq[128];
    __shared__ float s_amps[68];      // [b][k] occupancy softmax
    __shared__ float s_rm[34];        // row means of basis
    __shared__ float s_c[2];
    __shared__ float s_amp[768];      // resonance amp [b][i][d]
    __shared__ float s_smix[2304];    // [i][j][d]
    __shared__ float s_M[768];        // [b][j][d]
    __shared__ float s_tab[128];
    __shared__ float s_red[256];
    const int MN[6]   = {2, 3, 5, 8, 8, 8};
    const int KOFF[6] = {0, 2, 5, 10, 18, 26};
    int t = threadIdx.x;

    if (t < 128) s_seq[t] = p.seq[t];
    __syncthreads();

    // occupancy logits: 68 dot products of length 64
    if (t < 68) {
        int b = t / 34, k = t % 34;
        int i = well_of_k(k);
        int n = k - KOFF[i];
        float acc = p.bocc[i][n];
        for (int d = 0; d < D_DIM; d++)
            acc += s_seq[b * D_DIM + d] * p.wocc[i][n * D_DIM + d];
        s_amps[t] = acc;
    }
    __syncthreads();

    // softmax per (b, well)
    if (t < 12) {
        int b = t / 6, i = t % 6, mn = MN[i], off = b * 34 + KOFF[i];
        float mx = -1e30f;
        for (int n = 0; n < mn; n++) mx = fmaxf(mx, s_amps[off + n]);
        float sum = 0.f;
        for (int n = 0; n < mn; n++) { s_amps[off + n] = expf(s_amps[off + n] - mx); sum += s_amps[off + n]; }
        for (int n = 0; n < mn; n++) s_amps[off + n] /= sum;
    }
    // basis row means (independent threads)
    if (t < 34) {
        const float* row = p.basis + t * S_LEN;
        float acc = 0.f;
        for (int s = 0; s < S_LEN; s++) acc += row[s];
        s_rm[t] = acc / (float)S_LEN;
    }
    __syncthreads();

    if (t < 2) {
        float acc = 0.f;
        for (int k = 0; k < 34; k++) acc += s_amps[t * 34 + k] * s_rm[k];
        s_c[t] = acc / 6.0f;     // qE[b,d] = s_c[b] * seq_mean[b,d]
    }
    __syncthreads();

    // resonance amp [b,i,d] -> shared + output
    for (int u = t; u < 768; u += 256) {
        int b = u / 384, i = (u / 64) % 6, d = u % 64;
        float qE = s_c[b] * s_seq[b * D_DIM + d];
        float G  = fabsf(p.resW[i * D_DIM + d]);
        float df = qE - p.resE[i * D_DIM + d];
        float val = G / sqrtf(df * df + 0.25f * G * G + 1e-8f);
        s_amp[u] = val;
        p.out_amp[u] = val;
    }
    __syncthreads();

    // strength: mean over (b,d) of unbiased std over i
    if (t < 128) {
        int b = t / 64, d = t % 64;
        float m = 0.f;
        for (int i = 0; i < 6; i++) m += s_amp[(b * 6 + i) * 64 + d];
        m /= 6.0f;
        float v = 0.f;
        for (int i = 0; i < 6; i++) { float x = s_amp[(b * 6 + i) * 64 + d] - m; v += x * x; }
        s_red[t] = sqrtf(v / 5.0f);
    } else s_red[t] = 0.f;
    __syncthreads();
    for (int o = 128; o > 0; o >>= 1) { if (t < o) s_red[t] += s_red[t + o]; __syncthreads(); }
    if (t == 0) p.out_strength[0] = s_red[0] / 128.0f;

    // Smix = softmax_j(S_matrix / 0.1)
    for (int u = t; u < 384; u += 256) {
        int i = u / 64, d = u % 64;
        float vals[6]; float mx = -1e30f;
        for (int j = 0; j < 6; j++) { vals[j] = p.Smat[(i * 6 + j) * 64 + d] * 10.0f; mx = fmaxf(mx, vals[j]); }
        float sum = 0.f;
        for (int j = 0; j < 6; j++) { vals[j] = expf(vals[j] - mx); sum += vals[j]; }
        for (int j = 0; j < 6; j++) s_smix[(i * 6 + j) * 64 + d] = vals[j] / sum;
    }
    __syncthreads();

    // M[b,j,d] = sum_i amp[b,i,d] * Smix[i,j,d]
    for (int u = t; u < 768; u += 256) {
        int b = u / 384, j = (u / 64) % 6, d = u % 64;
        float acc = 0.f;
        for (int i = 0; i < 6; i++) acc += s_amp[(b * 6 + i) * 64 + d] * s_smix[(i * 6 + j) * 64 + d];
        s_M[u] = acc;
    }
    __syncthreads();

    // C[b,k,d] = amps[b,k] * M[b, well(k), d]  (to global ws)
    for (int u = t; u < 4352; u += 256) {
        int b = u / (34 * 64), k = (u / 64) % 34, d = u % 64;
        p.Cmat[u] = s_amps[b * 34 + k] * s_M[(b * 6 + well_of_k(k)) * 64 + d];
    }

    // tablet[b,d] = seq_mean * sum_k rowmean_k * C[b,k,d]  (recompute inline, full f32)
    if (t < 128) {
        int b = t / 64, d = t % 64;
        float acc = 0.f;
        for (int k = 0; k < 34; k++)
            acc += s_rm[k] * s_amps[b * 34 + k] * s_M[(b * 6 + well_of_k(k)) * 64 + d];
        float tab = s_seq[t] * acc;
        s_tab[t] = tab;
        p.out_tablet[t] = tab;
    }
    __syncthreads();

    if (t < 6) {
        int b = t / 3, g = t % 3;
        float acc = p.geob[g];
        for (int d = 0; d < D_DIM; d++) acc += s_tab[b * 64 + d] * p.geow[g * 64 + d];
        p.out_geo[b * 3 + g] = acc;
    }
}

// ---------------- final[m,d] = seq_mean[b,d] * sum_k basis[k,s]*C[b,k,d] -----
__global__ __launch_bounds__(256) void k_final(const float* __restrict__ seq,
                                               const float* __restrict__ basis,
                                               const float* __restrict__ Cmat,
                                               float* __restrict__ fin) {
    int g = blockIdx.x * 256 + threadIdx.x;   // 4096*64
    int m = g >> 6, d = g & 63;
    int b = m >> 11, s = m & 2047;
    float acc = 0.f;
    #pragma unroll 2
    for (int k = 0; k < 34; k++)
        acc += basis[k * S_LEN + s] * Cmat[(b * 34 + k) * 64 + d];
    fin[g] = seq[b * 64 + d] * acc;
}

// ---------------- pack A (final) into 16x16x32 f16 WMMA lane layout ----------
// lane<16: elems 0..7 -> K 0..7,  8..15 -> K 16..23 (plus 32*h)
// lane>=16: elems 0..7 -> K 8..15, 8..15 -> K 24..31 (plus 32*h)
__global__ void k_apack(const float* __restrict__ fin, _Float16* __restrict__ apack) {
    int g = blockIdx.x * 256 + threadIdx.x;   // 256 mtiles * 2 halves * 32 lanes
    int lane = g & 31, h = (g >> 5) & 1, mt = g >> 6;
    int row = mt * 16 + (lane & 15);
    _Float16* out = apack + ((size_t)(mt * 2 + h) * 32 + lane) * 16;
    int kb = 32 * h + 8 * (lane >> 4);
    #pragma unroll
    for (int e = 0; e < 16; e++) {
        int k = kb + 16 * (e >> 3) + (e & 7);
        out[e] = (_Float16)fin[row * 64 + k];
    }
}

// ---------------- pack B (sign_w, used transposed) into WMMA lane layout -----
// B is 32x16 (KxN): lane<16 elems 0..7 -> K 0..7, 8..15 -> K 16..23; lane>=16 elems 0..7 -> K 8..15, 8..15 -> K 24..31; N = lane%16
__global__ void k_bpack(const float* __restrict__ sw, _Float16* __restrict__ bpack) {
    int g = blockIdx.x * 256 + threadIdx.x;   // 2000 ntiles * 2 halves * 32 lanes
    int lane = g & 31, h = (g >> 5) & 1, nt = g >> 6;
    int v = nt * 16 + (lane & 15);
    int kb = 32 * h + 8 * (lane >> 4);
    _Float16* out = bpack + ((size_t)(nt * 2 + h) * 32 + lane) * 16;
    #pragma unroll
    for (int e = 0; e < 16; e++)
        out[e] = (_Float16)sw[v * 64 + kb + 16 * (e >> 3) + (e & 7)];
}

// ---------------- big GEMM: logits[4096,32000] = final @ sign_w.T + bias -----
// grid (250, 256); 8 waves/block; each wave one 16x16 tile; 2 WMMAs (K=64)
__global__ __launch_bounds__(256) void k_gemm(const _Float16* __restrict__ apack,
                                              const _Float16* __restrict__ bpack,
                                              const float* __restrict__ bias,
                                              float* __restrict__ out) {
    __shared__ float tile[16 * 128];
    int mt  = blockIdx.y;
    int nsb = blockIdx.x;
    int wave = threadIdx.x >> 5, lane = threadIdx.x & 31;
    int nt = nsb * 8 + wave;

    const v16h* ap = (const v16h*)(apack + (size_t)mt * 2 * 32 * 16);
    const v16h* bp = (const v16h*)(bpack + (size_t)nt * 2 * 32 * 16);
    v16h a0 = ap[lane],      a1 = ap[32 + lane];
    v16h b0 = bp[lane],      b1 = bp[32 + lane];

    v8f c = {};
    c = __builtin_amdgcn_wmma_f32_16x16x32_f16(false, a0, false, b0, (short)0, c, false, false);
    c = __builtin_amdgcn_wmma_f32_16x16x32_f16(false, a1, false, b1, (short)0, c, false, false);

    float bv = bias[nt * 16 + (lane & 15)];
    int col   = wave * 16 + (lane & 15);
    int rbase = (lane >> 4) * 8;                 // C layout: M = r + 8*(lane/16)
    #pragma unroll
    for (int r = 0; r < 8; r++)
        tile[(rbase + r) * 128 + col] = c[r] + bv;
    __syncthreads();

    // coalesced store: 16 rows x 128 cols as float4
    const float4* tp = (const float4*)tile;
    for (int i = threadIdx.x; i < 512; i += 256) {
        int row = i >> 5, c4 = i & 31;
        size_t m = (size_t)mt * 16 + row;
        *(float4*)(out + m * (size_t)V_SZ + (size_t)nsb * 128 + c4 * 4) = tp[i];
    }
}

extern "C" void kernel_launch(void* const* d_in, const int* in_sizes, int n_in,
                              void* d_out, int out_size, void* d_ws, size_t ws_size,
                              hipStream_t stream) {
    const int*   sign_ids  = (const int*)d_in[0];
    const float* embedding = (const float*)d_in[1];
    const float* wocc[6], *bocc[6];
    for (int i = 0; i < 6; i++) {
        wocc[i] = (const float*)d_in[2 + 2 * i];
        bocc[i] = (const float*)d_in[3 + 2 * i];
    }
    const float* Smat = (const float*)d_in[14];
    const float* resE = (const float*)d_in[15];
    const float* resW = (const float*)d_in[16];
    const float* sw   = (const float*)d_in[17];
    const float* sb   = (const float*)d_in[18];
    const float* geow = (const float*)d_in[19];
    const float* geob = (const float*)d_in[20];

    // output tuple layout: tablet[128] | logits[131072000] | geometry[6] | amp[768] | strength[1]
    float* out        = (float*)d_out;
    float* out_tab    = out;
    float* out_logits = out + 128;
    float* out_geo    = out + 128 + (long long)4096 * V_SZ;
    float* out_amp    = out_geo + 6;
    float* out_str    = out_amp + 768;

    // workspace layout (bytes)
    char* ws = (char*)d_ws;
    float*    seq   = (float*)(ws + 0);          // 128 f32
    float*    basis = (float*)(ws + 512);        // 34*2048 f32  -> 279040
    float*    Cmat  = (float*)(ws + 279040);     // 2*34*64 f32  -> 296448
    float*    fin   = (float*)(ws + 296448);     // 4096*64 f32  -> 1345024
    _Float16* apack = (_Float16*)(ws + 1345024); // 256*2*512 f16 -> 1869312
    _Float16* bpack = (_Float16*)(ws + 1869312); // 2000*2*512 f16 -> 5965312

    k_seqmean<<<2, 64, 0, stream>>>(sign_ids, embedding, seq);
    k_basis<<<6, 256, 0, stream>>>(basis);

    SmallParams p;
    p.seq = seq; p.basis = basis; p.Cmat = Cmat;
    for (int i = 0; i < 6; i++) { p.wocc[i] = wocc[i]; p.bocc[i] = bocc[i]; }
    p.Smat = Smat; p.resE = resE; p.resW = resW; p.geow = geow; p.geob = geob;
    p.out_tablet = out_tab; p.out_geo = out_geo; p.out_amp = out_amp; p.out_strength = out_str;
    k_small<<<1, 256, 0, stream>>>(p);

    k_final<<<1024, 256, 0, stream>>>(seq, basis, Cmat, fin);
    k_apack<<<64, 256, 0, stream>>>(fin, apack);
    k_bpack<<<500, 256, 0, stream>>>(sw, bpack);

    dim3 gg(250, 256, 1);
    k_gemm<<<gg, 256, 0, stream>>>(apack, bpack, sb, out_logits);
}
